// CostAttention_67817533604053
// MI455X (gfx1250) — matrix operand order repaired and checked by reading, not compiled, not run on hardware
//
#include <hip/hip_runtime.h>
#include <hip/hip_bf16.h>
#include <math.h>

typedef _Float16 half_t;
typedef __attribute__((ext_vector_type(16))) _Float16 v16h;
typedef __attribute__((ext_vector_type(8)))  _Float16 v8h;
typedef __attribute__((ext_vector_type(8)))  float    v8f;
typedef __attribute__((ext_vector_type(4)))  int      v4i_t;

#define B_  4
#define C_  64
#define N_  6400          // 80*80
#define QT  64            // query rows per block (4 waves x 16)
#define KT  64            // keys per main-loop iteration (double-buffered)
#define KTILES (N_ / KT)  // 100
#define CP  72            // padded row stride (halfs) for LDS tiles (144B, 16B-aligned)

// ---- address-space-typed pointers for async direct-to-LDS b128 loads ----
typedef __attribute__((address_space(1))) v4i_t gv4i_t;  // global int32x4
typedef __attribute__((address_space(3))) v4i_t lv4i_t;  // LDS    int32x4
#define TO_GLOBAL(p) ((gv4i_t*)(uintptr_t)(p))
#define TO_LDS(p)    ((lv4i_t*)(uint32_t)(uintptr_t)(p))

#if __has_builtin(__builtin_amdgcn_global_load_async_to_lds_b128)
#define HAVE_ASYNC_LDS 1
#else
#define HAVE_ASYNC_LDS 0
#endif

__device__ __forceinline__ void wait_asynccnt0() {
#if HAVE_ASYNC_LDS
#if __has_builtin(__builtin_amdgcn_s_wait_asynccnt)
    __builtin_amdgcn_s_wait_asynccnt(0);
#else
    asm volatile("s_wait_asynccnt 0" ::: "memory");
#endif
#endif
}

// ---- 16-lane butterfly reductions via DPP row_xmask (no ds traffic) ----
#if __has_builtin(__builtin_amdgcn_mov_dpp)
#define DPP_XOR(x, m) \
    __int_as_float(__builtin_amdgcn_mov_dpp(__float_as_int(x), 0x160 | (m), 0xF, 0xF, true))
__device__ __forceinline__ float half_row_max(float x) {
    x = fmaxf(x, DPP_XOR(x, 1));
    x = fmaxf(x, DPP_XOR(x, 2));
    x = fmaxf(x, DPP_XOR(x, 4));
    x = fmaxf(x, DPP_XOR(x, 8));
    return x;
}
__device__ __forceinline__ float half_row_sum(float x) {
    x += DPP_XOR(x, 1);
    x += DPP_XOR(x, 2);
    x += DPP_XOR(x, 4);
    x += DPP_XOR(x, 8);
    return x;
}
#else
__device__ __forceinline__ float half_row_max(float x) {
    for (int m = 1; m < 16; m <<= 1) x = fmaxf(x, __shfl_xor(x, m, 32));
    return x;
}
__device__ __forceinline__ float half_row_sum(float x) {
    for (int m = 1; m < 16; m <<= 1) x += __shfl_xor(x, m, 32);
    return x;
}
#endif

// ---- fragment loaders: force aligned b128 LDS reads ----
__device__ __forceinline__ v16h ldb16(const half_t* p) {   // 16 contiguous halfs
    const v8h* q = (const v8h*)p;
    v8h lo = q[0], hi = q[1];
    return __builtin_shufflevector(lo, hi, 0,1,2,3,4,5,6,7,8,9,10,11,12,13,14,15);
}
__device__ __forceinline__ v16h lda(const half_t* rowp, int kbase) {  // A layout: runs at kbase, kbase+8
    const v8h* lo = (const v8h*)(rowp + kbase);
    const v8h* hi = (const v8h*)(rowp + kbase + 8);
    return __builtin_shufflevector(*lo, *hi, 0,1,2,3,4,5,6,7,8,9,10,11,12,13,14,15);
}

// -------------------------------------------------------------------------
// Prep: qh = (Wq @ query + bq) * C^-0.5  f16, layout [B][N][C]
//       kh = keys^T f16, layout [B][N][C]
//       vh = values  f16, layout [B][C][N]  (plain cast, no transpose)
// -------------------------------------------------------------------------
__global__ void __launch_bounds__(256)
ca_prep_kernel(const float* __restrict__ query,
               const float* __restrict__ keys,
               const float* __restrict__ values,
               const float* __restrict__ Wq,
               const float* __restrict__ bq,
               half_t* __restrict__ qh,
               half_t* __restrict__ kh,
               half_t* __restrict__ vh)
{
    __shared__ float wqs[64 * 65];   // padded: row d at d*65
    __shared__ float tile[64 * 65];  // [c][n], padded
    __shared__ float bqs[64];

    const int tid = threadIdx.x;               // 0..255
    const int b   = blockIdx.y;
    const int n0  = blockIdx.x * 64;

    for (int i = tid; i < 64 * 64; i += 256)
        wqs[(i >> 6) * 65 + (i & 63)] = Wq[i];
    if (tid < 64) bqs[tid] = bq[tid];

    // ---- V: plain elementwise cast, coalesced ----
    for (int i = tid; i < 64 * 64; i += 256) {
        int c = i >> 6, n = i & 63;
        size_t idx = ((size_t)b * C_ + c) * N_ + n0 + n;
        vh[idx] = (half_t)values[idx];
    }

    // ---- query tile [c][n] ----
    for (int i = tid; i < 64 * 64; i += 256) {
        int c = i >> 6, n = i & 63;
        tile[c * 65 + n] = query[((size_t)b * C_ + c) * N_ + n0 + n];
    }
    __syncthreads();

    const int d  = tid & 63;
    const int nb = tid >> 6;
    const float scale = 0.125f;  // 64^-0.5 folded into Q

    for (int n = nb * 16; n < nb * 16 + 16; ++n) {
        float acc = bqs[d];
        #pragma unroll 8
        for (int c = 0; c < 64; ++c)
            acc = fmaf(wqs[d * 65 + c], tile[c * 65 + n], acc);
        qh[((size_t)b * N_ + n0 + n) * C_ + d] = (half_t)(acc * scale);
    }

    // ---- keys: transpose-cast to [N][C] ----
    __syncthreads();
    for (int i = tid; i < 64 * 64; i += 256) {
        int c = i >> 6, n = i & 63;
        tile[c * 65 + n] = keys[((size_t)b * C_ + c) * N_ + n0 + n];
    }
    __syncthreads();
    for (int n = nb * 16; n < nb * 16 + 16; ++n)
        kh[((size_t)b * N_ + n0 + n) * C_ + d] = (half_t)tile[d * 65 + n];
}

// -------------------------------------------------------------------------
// Flash attention, double-buffered async K/V staging.
//   Ks[buf][key][c]  (K rows, from kh [N][C])
//   Vt[buf][c][key]  (V cols, from vh [C][N])  -> contiguous B-fragments
// -------------------------------------------------------------------------
__global__ void __launch_bounds__(128)
ca_flash_kernel(const half_t* __restrict__ qh,
                const half_t* __restrict__ kh,
                const half_t* __restrict__ vh,
                float* __restrict__ out)
{
    __shared__ __align__(16) half_t Qs[QT][CP];
    __shared__ __align__(16) half_t Ks[2][KT][CP];
    __shared__ __align__(16) half_t Vt[2][C_][CP];
    __shared__ __align__(16) half_t Ps[4][16][CP];
    __shared__ float Os[QT][C_ + 1];

    const int tid  = threadIdx.x;     // 0..127
    const int wave = tid >> 5;
    const int lane = tid & 31;
    const int b    = blockIdx.y;
    const int n0   = blockIdx.x * QT;

    const half_t* kbase = kh + (size_t)b * N_ * C_;   // [key][c]
    const half_t* vbase = vh + (size_t)b * C_ * N_;   // [c][n]

    // issue (or copy) one K/V tile into LDS buffer `buf`
    auto stage_tiles = [&](int buf, int kt) {
        const half_t* ksrc = kbase + (size_t)kt * KT * C_;   // 64 rows x 64 halfs
        const half_t* vsrc = vbase + (size_t)kt * KT;        // 64 rows, stride N_
        for (int i = tid; i < KT * C_ / 8; i += 128) {       // 512 b128 per tensor
            int r = i >> 3, q = i & 7;
#if HAVE_ASYNC_LDS
            __builtin_amdgcn_global_load_async_to_lds_b128(
                TO_GLOBAL(ksrc + (size_t)r * C_ + q * 8), TO_LDS(&Ks[buf][r][q * 8]), 0, 0);
            __builtin_amdgcn_global_load_async_to_lds_b128(
                TO_GLOBAL(vsrc + (size_t)r * N_ + q * 8), TO_LDS(&Vt[buf][r][q * 8]), 0, 0);
#else
            *(uint4*)&Ks[buf][r][q * 8] = *(const uint4*)(ksrc + (size_t)r * C_ + q * 8);
            *(uint4*)&Vt[buf][r][q * 8] = *(const uint4*)(vsrc + (size_t)r * N_ + q * 8);
#endif
        }
    };

    // ---- stage Q tile (64x64 f16) into LDS + prologue tile 0 ----
    {
        const uint4* src = (const uint4*)(qh + ((size_t)b * N_ + n0) * C_);
        for (int i = tid; i < QT * C_ / 8; i += 128) {
            int r = i >> 3, q = i & 7;
            *(uint4*)&Qs[r][q * 8] = src[i];
        }
    }
    stage_tiles(0, 0);
    __syncthreads();

    // ---- per-wave fragment coordinates ----
    const int row    = lane & 15;            // M index
    const int col    = lane & 15;            // N index for B/C/D
    const int kbaseA = (lane >> 4) << 4;     // A layout: 0 or 16
    const int koffB  = (lane >> 4) << 4;     // B layout: 0 or 16
    const int drow0  = (lane >> 4) << 3;     // C/D layout: rows r or r+8

    const v16h aq0 = lda(&Qs[wave * 16 + row][0], kbaseA);
    const v16h aq1 = lda(&Qs[wave * 16 + row][32], kbaseA);

    v8f o0 = {}, o1 = {}, o2 = {}, o3 = {};
    float m[8], l[8];
    #pragma unroll
    for (int r = 0; r < 8; ++r) { m[r] = -INFINITY; l[r] = 0.0f; }

    for (int kt = 0; kt < KTILES; ++kt) {
        const int cur = kt & 1;
        wait_asynccnt0();      // my share of tile kt has landed
        __syncthreads();       // everyone's share landed; prev compute done
        if (kt + 1 < KTILES) stage_tiles(cur ^ 1, kt + 1);  // overlap DMA with compute

        // ---- S = Q * K^T : 4 key-column tiles x 2 chained WMMAs ----
        v8f s0 = {}, s1 = {}, s2 = {}, s3 = {};
        {
            const half_t* kr0 = &Ks[cur][col][koffB];
            const half_t* kr1 = &Ks[cur][16 + col][koffB];
            const half_t* kr2 = &Ks[cur][32 + col][koffB];
            const half_t* kr3 = &Ks[cur][48 + col][koffB];
            s0 = __builtin_amdgcn_wmma_f32_16x16x32_f16(false, aq0, false, ldb16(kr0), (short)0, s0, false, false);
            s0 = __builtin_amdgcn_wmma_f32_16x16x32_f16(false, aq1, false, ldb16(kr0 + 32), (short)0, s0, false, false);
            s1 = __builtin_amdgcn_wmma_f32_16x16x32_f16(false, aq0, false, ldb16(kr1), (short)0, s1, false, false);
            s1 = __builtin_amdgcn_wmma_f32_16x16x32_f16(false, aq1, false, ldb16(kr1 + 32), (short)0, s1, false, false);
            s2 = __builtin_amdgcn_wmma_f32_16x16x32_f16(false, aq0, false, ldb16(kr2), (short)0, s2, false, false);
            s2 = __builtin_amdgcn_wmma_f32_16x16x32_f16(false, aq1, false, ldb16(kr2 + 32), (short)0, s2, false, false);
            s3 = __builtin_amdgcn_wmma_f32_16x16x32_f16(false, aq0, false, ldb16(kr3), (short)0, s3, false, false);
            s3 = __builtin_amdgcn_wmma_f32_16x16x32_f16(false, aq1, false, ldb16(kr3 + 32), (short)0, s3, false, false);
        }

        // ---- online softmax over 64 keys ----
        float mt[8];
        #pragma unroll
        for (int r = 0; r < 8; ++r)
            mt[r] = half_row_max(fmaxf(fmaxf(s0[r], s1[r]), fmaxf(s2[r], s3[r])));

        float rs[8];
        #pragma unroll
        for (int r = 0; r < 8; ++r) {
            float nm   = fmaxf(m[r], mt[r]);
            float corr = __expf(m[r] - nm);
            m[r]  = nm;
            l[r] *= corr;
            o0[r] *= corr; o1[r] *= corr; o2[r] *= corr; o3[r] *= corr;
            float p0 = __expf(s0[r] - nm);
            float p1 = __expf(s1[r] - nm);
            float p2 = __expf(s2[r] - nm);
            float p3 = __expf(s3[r] - nm);
            rs[r] = (p0 + p1) + (p2 + p3);
            Ps[wave][drow0 + r][col]      = (half_t)p0;
            Ps[wave][drow0 + r][16 + col] = (half_t)p1;
            Ps[wave][drow0 + r][32 + col] = (half_t)p2;
            Ps[wave][drow0 + r][48 + col] = (half_t)p3;
        }
        #pragma unroll
        for (int r = 0; r < 8; ++r)
            l[r] += half_row_sum(rs[r]);

        // ---- O += P * V : K-dim 64 (2 chained) x 4 output column tiles ----
        const v16h ap0 = lda(&Ps[wave][row][0], kbaseA);   // keys 0..31
        const v16h ap1 = lda(&Ps[wave][row][32], kbaseA);  // keys 32..63
        {
            const half_t* vr0 = &Vt[cur][col][koffB];
            const half_t* vr1 = &Vt[cur][16 + col][koffB];
            const half_t* vr2 = &Vt[cur][32 + col][koffB];
            const half_t* vr3 = &Vt[cur][48 + col][koffB];
            o0 = __builtin_amdgcn_wmma_f32_16x16x32_f16(false, ap0, false, ldb16(vr0), (short)0, o0, false, false);
            o0 = __builtin_amdgcn_wmma_f32_16x16x32_f16(false, ap1, false, ldb16(vr0 + 32), (short)0, o0, false, false);
            o1 = __builtin_amdgcn_wmma_f32_16x16x32_f16(false, ap0, false, ldb16(vr1), (short)0, o1, false, false);
            o1 = __builtin_amdgcn_wmma_f32_16x16x32_f16(false, ap1, false, ldb16(vr1 + 32), (short)0, o1, false, false);
            o2 = __builtin_amdgcn_wmma_f32_16x16x32_f16(false, ap0, false, ldb16(vr2), (short)0, o2, false, false);
            o2 = __builtin_amdgcn_wmma_f32_16x16x32_f16(false, ap1, false, ldb16(vr2 + 32), (short)0, o2, false, false);
            o3 = __builtin_amdgcn_wmma_f32_16x16x32_f16(false, ap0, false, ldb16(vr3), (short)0, o3, false, false);
            o3 = __builtin_amdgcn_wmma_f32_16x16x32_f16(false, ap1, false, ldb16(vr3 + 32), (short)0, o3, false, false);
        }
    }

    // ---- normalize and stage to LDS for coalesced transposed store ----
    #pragma unroll
    for (int r = 0; r < 8; ++r) {
        float inv = 1.0f / l[r];
        int rr = wave * 16 + drow0 + r;
        Os[rr][col]      = o0[r] * inv;
        Os[rr][16 + col] = o1[r] * inv;
        Os[rr][32 + col] = o2[r] * inv;
        Os[rr][48 + col] = o3[r] * inv;
    }
    __syncthreads();

    // out layout [B][C][N]; lanes consecutive in n -> coalesced 256B rows
    float* obase = out + (size_t)b * C_ * N_ + n0;
    for (int i = tid; i < QT * C_; i += 128) {
        int c = i >> 6, n = i & 63;
        obase[(size_t)c * N_ + n] = Os[n][c];
    }
}

// -------------------------------------------------------------------------
extern "C" void kernel_launch(void* const* d_in, const int* in_sizes, int n_in,
                              void* d_out, int out_size, void* d_ws, size_t ws_size,
                              hipStream_t stream) {
    const float* query  = (const float*)d_in[0];
    const float* keys   = (const float*)d_in[1];
    const float* values = (const float*)d_in[2];
    const float* Wq     = (const float*)d_in[3];
    const float* bq     = (const float*)d_in[4];
    float* out = (float*)d_out;

    const size_t per = (size_t)B_ * N_ * C_;   // 1,638,400 halfs each
    half_t* qh = (half_t*)d_ws;
    half_t* kh = qh + per;
    half_t* vh = kh + per;

    dim3 pgrid(N_ / 64, B_);
    ca_prep_kernel<<<pgrid, 256, 0, stream>>>(query, keys, values, Wq, bq, qh, kh, vh);

    dim3 fgrid(N_ / QT, B_);
    ca_flash_kernel<<<fgrid, 128, 0, stream>>>(qh, kh, vh, out);
}
